// learn2graph_867583394077
// MI455X (gfx1250) — compile-verified, matrix operand order checked by reading, not run on hardware
//
#include <hip/hip_runtime.h>
#include <math.h>

// ---------------- problem constants ----------------
constexpr int B_   = 256;
constexpr int M_   = 128;
constexpr int T_   = 20;
constexpr int L_   = M_ * (M_ - 1) / 2;          // 8128
constexpr int NH_  = 64;
constexpr int NL_  = 32;
constexpr int NG_  = 64;
constexpr int HD_  = 10837;                      // int(M*(M-1)*2/3)
constexpr int K1_  = L_ + NL_;                   // 8160 (latent dim), %32 == 0
constexpr int HDP_ = 10848;                      // HD_ padded to multiple of 32
constexpr float TH_  = 1e-4f;
constexpr float KLH_ = 1.0f;

// output layout (flat concat of return tuple)
constexpr size_t OUT_WLIST  = 0;                                   // [B,T,L]
constexpr size_t OUT_LOSS   = (size_t)B_ * T_ * L_;                // scalar
constexpr size_t OUT_KL     = OUT_LOSS + 1;                        // scalar
constexpr size_t OUT_LATENT = OUT_KL + 1;                          // [B, K1]

// workspace layout (float slots; ints share slots)
constexpr size_t OFF_EI    = 0;                       // L ints
constexpr size_t OFF_EJ    = OFF_EI + L_;             // L ints
constexpr size_t OFF_W     = OFF_EJ + L_;             // B*L
constexpr size_t OFF_V     = OFF_W + (size_t)B_ * L_; // B*M
constexpr size_t OFF_Y1    = OFF_V + (size_t)B_ * M_;
constexpr size_t OFF_P1    = OFF_Y1 + (size_t)B_ * L_;
constexpr size_t OFF_Y2    = OFF_P1 + (size_t)B_ * L_;
constexpr size_t OFF_P2    = OFF_Y2 + (size_t)B_ * M_;
constexpr size_t OFF_ADJ   = OFF_P2 + (size_t)B_ * M_;             // B*M*M
constexpr size_t OFF_DEG   = OFF_ADJ + (size_t)B_ * M_ * M_;
constexpr size_t OFF_S     = OFF_DEG + (size_t)B_ * M_;
constexpr size_t OFF_H1    = OFF_S + (size_t)B_ * M_;              // B*M*64
constexpr size_t OFF_T2    = OFF_H1 + (size_t)B_ * M_ * NH_;       // B*M*128
constexpr size_t OFF_HG    = OFF_T2 + (size_t)B_ * M_ * 2 * NH_;   // B*128
constexpr size_t OFF_HGA   = OFF_HG + (size_t)B_ * 2 * NH_;        // B*64
constexpr size_t OFF_EMBX  = OFF_HGA + (size_t)B_ * NH_;           // B*64
constexpr size_t OFF_EMBY  = OFF_EMBX + (size_t)B_ * NG_;
constexpr size_t OFF_ZM    = OFF_EMBY + (size_t)B_ * NG_;          // B*32
constexpr size_t OFF_ZLV   = OFF_ZM + (size_t)B_ * NL_;
constexpr size_t OFF_ABF   = OFF_ZLV + (size_t)B_ * NL_;           // B*K1 bf16 == B*K1/2 floats
constexpr size_t OFF_HBUF  = OFF_ABF + (size_t)B_ * K1_ / 2;       // B*HDP bf16 == B*HDP/2 floats
constexpr size_t OFF_YPRED = OFF_HBUF + (size_t)B_ * HDP_ / 2;     // B*L f32
constexpr size_t OFF_PART  = OFF_YPRED + (size_t)B_ * L_;          // 1024

// ---------------- types / helpers ----------------
typedef __attribute__((ext_vector_type(16))) __bf16 v16bf;
typedef __attribute__((ext_vector_type(8)))  float  v8f;

struct Frag32 { unsigned int u[8]; };   // 32 bytes, bit-castable to v16bf

__device__ inline unsigned short f32_to_bf16_bits(float f) {
  unsigned int u = __builtin_bit_cast(unsigned int, f);
  unsigned int r = (u + 0x7FFFu + ((u >> 16) & 1u)) >> 16;
  return (unsigned short)r;
}
__device__ inline __bf16 f2bf(float f) {
  return __builtin_bit_cast(__bf16, f32_to_bf16_bits(f));
}
__device__ inline v16bf ld_frag(const __bf16* base, int half) {
  Frag32 f;
  const unsigned int* p0 = reinterpret_cast<const unsigned int*>(base + 8 * half);
  const unsigned int* p1 = reinterpret_cast<const unsigned int*>(base + 16 + 8 * half);
#pragma unroll
  for (int q = 0; q < 4; ++q) { f.u[q] = p0[q]; f.u[4 + q] = p1[q]; }
  return __builtin_bit_cast(v16bf, f);
}
__device__ inline int edge_idx(int i, int j) {   // requires i<j
  return i * (2 * M_ - i - 1) / 2 + (j - i - 1);
}
__device__ inline float prox_lb(float y, float g, float a) {
  float up = y * y + 4.0f * g * a;
  up = fmaxf(up, 1e-8f);
  return (y - sqrtf(up)) * 0.5f;
}

// ---------------- setup kernels ----------------
__global__ void k_edges(int* ei, int* ej) {
  int e = blockIdx.x * blockDim.x + threadIdx.x;
  if (e >= L_) return;
  int rem = e, i = 0;
  while (rem >= (M_ - 1 - i)) { rem -= (M_ - 1 - i); ++i; }
  ei[e] = i; ej[e] = i + 1 + rem;
}
__global__ void k_zero(float* p, long n) {
  long i = (long)blockIdx.x * blockDim.x + threadIdx.x;
  if (i < n) p[i] = 0.0f;
}

// ---------------- primal-dual scan kernels ----------------
__global__ void k_step_edge1(const float* w, const float* v, const float* z,
                             const int* ei, const int* ej,
                             const float* gn, const float* beta, int t,
                             float* y1, float* p1) {
  long idx = (long)blockIdx.x * blockDim.x + threadIdx.x;
  if (idx >= (long)B_ * L_) return;
  int b = (int)(idx / L_), e = (int)(idx % L_);
  float g = gn[t], bb = beta[t];
  float vv = v[b * M_ + ei[e]] + v[b * M_ + ej[e]];
  float wi = w[idx];
  float y = wi - g * (2.0f * bb * wi + 2.0f * z[idx] + vv);
  y1[idx] = y;
  p1[idx] = (y >= 0.0f) ? y : 0.01f * y;   // leaky_relu
}
__global__ void k_step_node1(const float* w, const float* v,
                             const float* gn, const float* alpha, int t,
                             float* y2, float* p2) {
  int idx = blockIdx.x * blockDim.x + threadIdx.x;
  if (idx >= B_ * M_) return;
  int b = idx / M_, i = idx % M_;
  float g = gn[t], aa = alpha[t];
  float s = 0.0f;
  const float* wb = w + (long)b * L_;
  for (int j = 0; j < M_; ++j) {
    if (j == i) continue;
    int a = i < j ? i : j, c = i < j ? j : i;
    s += wb[edge_idx(a, c)];
  }
  float y = v[idx] + g * s;
  y2[idx] = y;
  p2[idx] = prox_lb(y, g, aa);
}
__global__ void k_step_edge2(float* w, const float* z, const float* y1,
                             const float* p1, const float* p2,
                             const int* ei, const int* ej,
                             const float* gn, const float* beta, int t,
                             float* out_wlist) {
  long idx = (long)blockIdx.x * blockDim.x + threadIdx.x;
  if (idx >= (long)B_ * L_) return;
  int b = (int)(idx / L_), e = (int)(idx % L_);
  float g = gn[t], bb = beta[t];
  float pp = p2[b * M_ + ei[e]] + p2[b * M_ + ej[e]];
  float p = p1[idx];
  float q1 = p - g * (2.0f * bb * p + 2.0f * z[idx] + pp);
  float wn = w[idx] - y1[idx] + q1;
  w[idx] = wn;
  out_wlist[(long)b * T_ * L_ + (long)t * L_ + e] = wn;
}
__global__ void k_step_node2(float* v, const float* y2, const float* p2,
                             const float* p1, const float* gn, int t) {
  int idx = blockIdx.x * blockDim.x + threadIdx.x;
  if (idx >= B_ * M_) return;
  int b = idx / M_, i = idx % M_;
  float g = gn[t];
  float s = 0.0f;
  const float* pb = p1 + (long)b * L_;
  for (int j = 0; j < M_; ++j) {
    if (j == i) continue;
    int a = i < j ? i : j, c = i < j ? j : i;
    s += pb[edge_idx(a, c)];
  }
  float q2 = p2[idx] + g * s;
  v[idx] = v[idx] - y2[idx] + q2;
}

// ---------------- GCN encoder kernels ----------------
__global__ void k_adj(const float* src, float* adj) {
  long idx = (long)blockIdx.x * blockDim.x + threadIdx.x;
  if (idx >= (long)B_ * M_ * M_) return;
  int b = (int)(idx / (M_ * M_));
  int r = (int)(idx % (M_ * M_));
  int i = r / M_, j = r % M_;
  float val = 0.0f;
  if (i != j) {
    int a = i < j ? i : j, c = i < j ? j : i;
    val = (src[(long)b * L_ + edge_idx(a, c)] > TH_) ? 1.0f : 0.0f;
  }
  adj[idx] = val;
}
__global__ void k_deg(const float* adj, float* deg) {
  int idx = blockIdx.x * blockDim.x + threadIdx.x;
  if (idx >= B_ * M_) return;
  const float* row = adj + (long)idx * M_;
  float s = 0.0f;
  for (int j = 0; j < M_; ++j) s += row[j];
  deg[idx] = s;
}
__global__ void k_nds(const float* adj, const float* deg, float* sbuf) {
  int idx = blockIdx.x * blockDim.x + threadIdx.x;
  if (idx >= B_ * M_) return;
  int b = idx / M_;
  const float* row = adj + (long)idx * M_;
  const float* db = deg + b * M_;
  float s = 0.0f;
  for (int j = 0; j < M_; ++j) s += row[j] * db[j];
  sbuf[idx] = s;
}
__global__ void k_h1(const float* sbuf, const float* c1w, const float* c1b, float* h1) {
  long idx = (long)blockIdx.x * blockDim.x + threadIdx.x;
  if (idx >= (long)B_ * M_ * NH_) return;
  int bm = (int)(idx / NH_), k = (int)(idx % NH_);
  float v = sbuf[bm] * c1w[k] + c1b[k];
  h1[idx] = fmaxf(v, 0.0f);
}
__global__ void k_t2(const float* h1, const float* c2w, float* t2) {
  long idx = (long)blockIdx.x * blockDim.x + threadIdx.x;
  if (idx >= (long)B_ * M_ * 2 * NH_) return;
  int bm = (int)(idx / (2 * NH_)), j = (int)(idx % (2 * NH_));
  const float* h = h1 + (long)bm * NH_;
  float s = 0.0f;
  for (int k = 0; k < NH_; ++k) s += h[k] * c2w[k * 2 * NH_ + j];
  t2[idx] = s;
}
__global__ void k_hg(const float* adj, const float* t2, const float* c2b, float* hg) {
  int idx = blockIdx.x * blockDim.x + threadIdx.x;
  if (idx >= B_ * 2 * NH_) return;
  int b = idx / (2 * NH_), j = idx % (2 * NH_);
  const float* ab = adj + (long)b * M_ * M_;
  const float* tb = t2 + (long)b * M_ * 2 * NH_;
  float bias = c2b[j];
  float tot = 0.0f;
  for (int m = 0; m < M_; ++m) {
    const float* row = ab + (long)m * M_;
    float acc = 0.0f;
    for (int n = 0; n < M_; ++n) acc += row[n] * tb[(long)n * 2 * NH_ + j];
    tot += fmaxf(acc + bias, 0.0f);
  }
  hg[idx] = tot;
}
__global__ void k_fc1(const float* hg, const float* w, const float* bias, float* out) {
  int idx = blockIdx.x * blockDim.x + threadIdx.x;
  if (idx >= B_ * NH_) return;
  int b = idx / NH_, o = idx % NH_;
  const float* h = hg + (long)b * 2 * NH_;
  float s = bias[o];
  for (int j = 0; j < 2 * NH_; ++j) s += h[j] * w[j * NH_ + o];
  out[idx] = fmaxf(s, 0.0f);
}
__global__ void k_fc2(const float* hga, const float* w, const float* bias, float* emb) {
  int idx = blockIdx.x * blockDim.x + threadIdx.x;
  if (idx >= B_ * NG_) return;
  int b = idx / NG_, o = idx % NG_;
  const float* h = hga + (long)b * NH_;
  float s = bias[o];
  for (int j = 0; j < NH_; ++j) s += h[j] * w[j * NG_ + o];
  emb[idx] = s;
}

// ---------------- VAE latent kernels ----------------
__global__ void k_copy_y1_latent(const float* y1, float* out_latent) {
  long idx = (long)blockIdx.x * blockDim.x + threadIdx.x;
  if (idx >= (long)B_ * L_) return;
  int b = (int)(idx / L_), e = (int)(idx % L_);
  out_latent[(long)b * K1_ + e] = y1[idx];
}
__global__ void k_zstats(const float* embx, const float* emby,
                         const float* fmw, const float* fmb,
                         const float* fvw, const float* fvb,
                         const float* eps, float* zm, float* zlv, float* out_latent) {
  int idx = blockIdx.x * blockDim.x + threadIdx.x;
  if (idx >= B_ * NL_) return;
  int b = idx / NL_, o = idx % NL_;
  float sm = 0.0f, sv = 0.0f;
  for (int g = 0; g < NG_; ++g) {
    float d = emby[b * NG_ + g] - embx[b * NG_ + g];
    sm += d * fmw[g * NL_ + o];
    sv += d * fvw[g * NL_ + o];
  }
  float m = sm + fmb[o];
  float lv = -fabsf(sv + fvb[o]);
  zm[idx] = m;
  zlv[idx] = lv;
  out_latent[(long)b * K1_ + L_ + o] = m + expf(0.5f * lv) * eps[idx];
}
__global__ void k_kl(const float* zm, const float* zlv, float* out_kl) {
  __shared__ float red[256];
  int tid = threadIdx.x;
  float s = 0.0f;
  for (int i = tid; i < B_ * NL_; i += 256) {
    float m = zm[i], lv = zlv[i];
    s += 1.0f + lv - m * m - expf(lv);
  }
  red[tid] = s;
  __syncthreads();
  for (int st = 128; st > 0; st >>= 1) {
    if (tid < st) red[tid] += red[tid + st];
    __syncthreads();
  }
  if (tid == 0) out_kl[0] = -0.5f * red[0] / (float)(B_ * NL_);
}
// one-time latent f32 -> bf16 (removes per-block conversion from GEMM1 hot loop)
__global__ void k_lat_bf16(const float* latent, __bf16* abf) {
  long idx = (long)blockIdx.x * blockDim.x + threadIdx.x;
  if (idx >= (long)B_ * K1_) return;
  abf[idx] = f2bf(latent[idx]);
}

// ---------------- WMMA decoder GEMM (unified) ----------------
// C[256, N] = epilogue(A_bf16[256, LDA] @ W_f32[KTRUE, LDW] + bias)
// A tiles staged Global->LDS with CDNA5 async copies (ASYNCcnt path);
// W tiles converted f32->bf16 through VGPRs into LDS (format change).
// EPI==0: tanh -> bf16 out (stride LDO, zero-fill cols in [NTRUE, LDO))
// EPI==1: plain -> f32 out (stride LDO)
template <int KTOT, int LDA, int LDW, int KTRUE, int NTRUE, int EPI, int LDO>
__global__ void gemm_wmma(const __bf16* __restrict__ A,
                          const float* __restrict__ W,
                          const float* __restrict__ bias,
                          void* __restrict__ OutV) {
  __shared__ __align__(16) __bf16 ldsA[64 * 32];
  __shared__ __align__(16) __bf16 ldsB[64 * 32];   // stored [n_local][k_local]
  const int tid = threadIdx.x;
  const int n0 = blockIdx.x * 64;
  const int m0 = blockIdx.y * 64;
  const int wave = tid >> 5, lane = tid & 31;
  const int half = lane >> 4, r = lane & 15;
  const int wr = (wave >> 1) * 32, wc = (wave & 1) * 32;

  v8f acc[2][2];
#pragma unroll
  for (int s = 0; s < 2; ++s)
#pragma unroll
    for (int t = 0; t < 2; ++t)
#pragma unroll
      for (int q = 0; q < 8; ++q) acc[s][t][q] = 0.0f;

  // A staging: thread -> (row, 32B half); 64 rows x 64B per K-chunk
  const int arow = tid >> 1, ac = tid & 1;
  // LDS byte address (generic LDS pointers carry the LDS offset in bits[31:0])
  const unsigned ldsA_addr =
      (unsigned)(uintptr_t)(const void*)&ldsA[arow * 32 + ac * 16];
  // B staging: thread -> (n_local, 16 k values)
  const int bn = tid >> 1, bk0 = (tid & 1) * 16;

  for (int kk = 0; kk < KTOT; kk += 32) {
    // ---- async Global->LDS copy of A tile (2 x b128 per thread) ----
    {
      unsigned long long ga = (unsigned long long)(uintptr_t)(
          A + (long)(m0 + arow) * LDA + kk + ac * 16);
      asm volatile(
          "global_load_async_to_lds_b128 %0, %1, off\n\t"
          "global_load_async_to_lds_b128 %0, %1, off offset:16"
          :: "v"(ldsA_addr), "v"(ga) : "memory");
    }
    // ---- W tile: f32 -> bf16 via VGPRs, transposed into LDS [n][k] ----
    {
      int n = n0 + bn;
      __bf16* dst = &ldsB[bn * 32 + bk0];
      if (n < NTRUE) {
        const float* src = W + (long)(kk + bk0) * LDW + n;
#pragma unroll
        for (int q = 0; q < 16; ++q) {
          int k = kk + bk0 + q;
          float val = (k < KTRUE) ? src[(long)q * LDW] : 0.0f;
          dst[q] = f2bf(val);
        }
      } else {
#pragma unroll
        for (int q = 0; q < 16; ++q) dst[q] = f2bf(0.0f);
      }
    }
    asm volatile("s_wait_asynccnt 0" ::: "memory");  // A tile landed in LDS
    __syncthreads();                                 // covers B-tile ds stores

    v16bf afr[2], bfr[2];
#pragma unroll
    for (int s = 0; s < 2; ++s) {
      afr[s] = ld_frag(&ldsA[(wr + 16 * s + r) * 32], half);
      bfr[s] = ld_frag(&ldsB[(wc + 16 * s + r) * 32], half);
    }
#pragma unroll
    for (int s = 0; s < 2; ++s)
#pragma unroll
      for (int t = 0; t < 2; ++t)
        acc[s][t] = __builtin_amdgcn_wmma_f32_16x16x32_bf16(
            false, afr[s], false, bfr[t], (short)0, acc[s][t], false, false);
    __syncthreads();
  }

#pragma unroll
  for (int s = 0; s < 2; ++s)
#pragma unroll
    for (int t = 0; t < 2; ++t) {
      int col = n0 + wc + 16 * t + r;
      if (EPI == 0) {
        __bf16* Out = (__bf16*)OutV;
        if (col < LDO) {
          float bv = (col < NTRUE) ? bias[col] : 0.0f;
#pragma unroll
          for (int q = 0; q < 8; ++q) {
            int row = m0 + wr + 16 * s + q + 8 * half;
            float val = (col < NTRUE) ? tanhf(acc[s][t][q] + bv) : 0.0f;
            Out[(long)row * LDO + col] = f2bf(val);
          }
        }
      } else {
        float* Out = (float*)OutV;
        if (col < NTRUE) {
          float bv = bias[col];
#pragma unroll
          for (int q = 0; q < 8; ++q) {
            int row = m0 + wr + 16 * s + q + 8 * half;
            Out[(long)row * LDO + col] = acc[s][t][q] + bv;
          }
        }
      }
    }
}

// ---------------- loss + final step kernels ----------------
__global__ void k_recons_part(const float* y_pred, const float* w_gt, float* partial) {
  __shared__ float red[256];
  int tid = threadIdx.x;
  long n = (long)B_ * L_;
  float s = 0.0f;
  for (long i = (long)blockIdx.x * 256 + tid; i < n; i += (long)gridDim.x * 256) {
    float d = y_pred[i] - w_gt[i];
    s += d * d;
  }
  red[tid] = s;
  __syncthreads();
  for (int st = 128; st > 0; st >>= 1) {
    if (tid < st) red[tid] += red[tid + st];
    __syncthreads();
  }
  if (tid == 0) partial[blockIdx.x] = red[0];
}
__global__ void k_loss_final(const float* partial, const float* kl_ptr, float* out_loss) {
  __shared__ float red[256];
  int tid = threadIdx.x;
  float s = 0.0f;
  for (int i = tid; i < 1024; i += 256) s += partial[i];
  red[tid] = s;
  __syncthreads();
  for (int st = 128; st > 0; st >>= 1) {
    if (tid < st) red[tid] += red[tid + st];
    __syncthreads();
  }
  if (tid == 0) out_loss[0] = red[0] / (float)B_ + KLH_ * kl_ptr[0];
}
__global__ void k_final_w(const float* w, const float* y1, const float* p2,
                          const float* y_pred, const float* z,
                          const int* ei, const int* ej,
                          const float* gn, const float* beta, float* out_wlist) {
  long idx = (long)blockIdx.x * blockDim.x + threadIdx.x;
  if (idx >= (long)B_ * L_) return;
  int b = (int)(idx / L_), e = (int)(idx % L_);
  float g = gn[T_ - 1], bb = beta[T_ - 1];
  float p1v = y_pred[idx];
  float pp = p2[b * M_ + ei[e]] + p2[b * M_ + ej[e]];
  float q1 = p1v - g * (2.0f * bb * p1v + 2.0f * z[idx] + pp);
  float wn = w[idx] - y1[idx] + q1;
  out_wlist[(long)b * T_ * L_ + (long)(T_ - 1) * L_ + e] = wn;
}

// ---------------- launch ----------------
extern "C" void kernel_launch(void* const* d_in, const int* in_sizes, int n_in,
                              void* d_out, int out_size, void* d_ws, size_t ws_size,
                              hipStream_t stream) {
  const float* z      = (const float*)d_in[0];
  const float* w_gt   = (const float*)d_in[1];
  const float* eps    = (const float*)d_in[2];
  const float* gn     = (const float*)d_in[3];
  const float* beta   = (const float*)d_in[4];
  const float* alpha  = (const float*)d_in[5];
  const float* c1w    = (const float*)d_in[6];
  const float* c1b    = (const float*)d_in[7];
  const float* c2w    = (const float*)d_in[8];
  const float* c2b    = (const float*)d_in[9];
  const float* fc1w   = (const float*)d_in[10];
  const float* fc1b   = (const float*)d_in[11];
  const float* fc2w   = (const float*)d_in[12];
  const float* fc2b   = (const float*)d_in[13];
  const float* fmw    = (const float*)d_in[14];
  const float* fmb    = (const float*)d_in[15];
  const float* fvw    = (const float*)d_in[16];
  const float* fvb    = (const float*)d_in[17];
  const float* dec1w  = (const float*)d_in[18];
  const float* dec1b  = (const float*)d_in[19];
  const float* dec2w  = (const float*)d_in[20];
  const float* dec2b  = (const float*)d_in[21];

  float* out = (float*)d_out;
  float* wsf = (float*)d_ws;

  int*   ei    = (int*)(wsf + OFF_EI);
  int*   ej    = (int*)(wsf + OFF_EJ);
  float* w     = wsf + OFF_W;
  float* v     = wsf + OFF_V;
  float* y1    = wsf + OFF_Y1;
  float* p1    = wsf + OFF_P1;
  float* y2    = wsf + OFF_Y2;
  float* p2    = wsf + OFF_P2;
  float* adj   = wsf + OFF_ADJ;
  float* deg   = wsf + OFF_DEG;
  float* sbuf  = wsf + OFF_S;
  float* h1    = wsf + OFF_H1;
  float* t2    = wsf + OFF_T2;
  float* hg    = wsf + OFF_HG;
  float* hga   = wsf + OFF_HGA;
  float* embx  = wsf + OFF_EMBX;
  float* emby  = wsf + OFF_EMBY;
  float* zm    = wsf + OFF_ZM;
  float* zlv   = wsf + OFF_ZLV;
  __bf16* Abf  = (__bf16*)(wsf + OFF_ABF);
  __bf16* Hbuf = (__bf16*)(wsf + OFF_HBUF);
  float* ypred = wsf + OFF_YPRED;
  float* part  = wsf + OFF_PART;

  float* out_wlist  = out + OUT_WLIST;
  float* out_loss   = out + OUT_LOSS;
  float* out_kl     = out + OUT_KL;
  float* out_latent = out + OUT_LATENT;

  const int TPB = 256;
  const long nBL = (long)B_ * L_;
  const int  nBM = B_ * M_;
  const long nBMM = (long)B_ * M_ * M_;

  dim3 gBL((unsigned)((nBL + TPB - 1) / TPB));
  dim3 gBM((unsigned)((nBM + TPB - 1) / TPB));
  dim3 gBMM((unsigned)((nBMM + TPB - 1) / TPB));

  k_edges<<<dim3((L_ + TPB - 1) / TPB), TPB, 0, stream>>>(ei, ej);
  k_zero<<<dim3((unsigned)((nBL + nBM + TPB - 1) / TPB)), TPB, 0, stream>>>(w, nBL + nBM);

  // 19 scan steps
  for (int t = 0; t < T_ - 1; ++t) {
    k_step_edge1<<<gBL, TPB, 0, stream>>>(w, v, z, ei, ej, gn, beta, t, y1, p1);
    k_step_node1<<<gBM, TPB, 0, stream>>>(w, v, gn, alpha, t, y2, p2);
    k_step_edge2<<<gBL, TPB, 0, stream>>>(w, z, y1, p1, p2, ei, ej, gn, beta, t, out_wlist);
    k_step_node2<<<gBM, TPB, 0, stream>>>(v, y2, p2, p1, gn, t);
  }
  // final y1/y2/p2 (t = T-1)
  k_step_edge1<<<gBL, TPB, 0, stream>>>(w, v, z, ei, ej, gn, beta, T_ - 1, y1, p1);
  k_step_node1<<<gBM, TPB, 0, stream>>>(w, v, gn, alpha, T_ - 1, y2, p2);

  k_copy_y1_latent<<<gBL, TPB, 0, stream>>>(y1, out_latent);

  // graph encoder pass 1 (adj from y1) -> embx
  k_adj<<<gBMM, TPB, 0, stream>>>(y1, adj);
  k_deg<<<gBM, TPB, 0, stream>>>(adj, deg);
  k_nds<<<gBM, TPB, 0, stream>>>(adj, deg, sbuf);
  k_h1<<<dim3((unsigned)(((long)B_ * M_ * NH_ + TPB - 1) / TPB)), TPB, 0, stream>>>(sbuf, c1w, c1b, h1);
  k_t2<<<dim3((unsigned)(((long)B_ * M_ * 2 * NH_ + TPB - 1) / TPB)), TPB, 0, stream>>>(h1, c2w, t2);
  k_hg<<<dim3((B_ * 2 * NH_ + TPB - 1) / TPB), TPB, 0, stream>>>(adj, t2, c2b, hg);
  k_fc1<<<dim3((B_ * NH_ + TPB - 1) / TPB), TPB, 0, stream>>>(hg, fc1w, fc1b, hga);
  k_fc2<<<dim3((B_ * NG_ + TPB - 1) / TPB), TPB, 0, stream>>>(hga, fc2w, fc2b, embx);

  // graph encoder pass 2 (adj from w_gt) -> emby
  k_adj<<<gBMM, TPB, 0, stream>>>(w_gt, adj);
  k_deg<<<gBM, TPB, 0, stream>>>(adj, deg);
  k_nds<<<gBM, TPB, 0, stream>>>(adj, deg, sbuf);
  k_h1<<<dim3((unsigned)(((long)B_ * M_ * NH_ + TPB - 1) / TPB)), TPB, 0, stream>>>(sbuf, c1w, c1b, h1);
  k_t2<<<dim3((unsigned)(((long)B_ * M_ * 2 * NH_ + TPB - 1) / TPB)), TPB, 0, stream>>>(h1, c2w, t2);
  k_hg<<<dim3((B_ * 2 * NH_ + TPB - 1) / TPB), TPB, 0, stream>>>(adj, t2, c2b, hg);
  k_fc1<<<dim3((B_ * NH_ + TPB - 1) / TPB), TPB, 0, stream>>>(hg, fc1w, fc1b, hga);
  k_fc2<<<dim3((B_ * NG_ + TPB - 1) / TPB), TPB, 0, stream>>>(hga, fc2w, fc2b, emby);

  // latent stats + kl
  k_zstats<<<dim3((B_ * NL_ + TPB - 1) / TPB), TPB, 0, stream>>>(
      embx, emby, fmw, fmb, fvw, fvb, eps, zm, zlv, out_latent);
  k_kl<<<dim3(1), TPB, 0, stream>>>(zm, zlv, out_kl);

  // latent -> bf16 once, then WMMA decoder GEMMs
  k_lat_bf16<<<dim3((unsigned)(((long)B_ * K1_ + TPB - 1) / TPB)), TPB, 0, stream>>>(
      out_latent, Abf);
  gemm_wmma<K1_, K1_, HD_, K1_, HD_, 0, HDP_>
      <<<dim3((HDP_ + 63) / 64, B_ / 64), 128, 0, stream>>>(Abf, dec1w, dec1b, (void*)Hbuf);
  gemm_wmma<HDP_, HDP_, L_, HD_, L_, 1, L_>
      <<<dim3(L_ / 64, B_ / 64), 128, 0, stream>>>(Hbuf, dec2w, dec2b, (void*)ypred);

  // reconstruction loss (deterministic two-stage reduction)
  k_recons_part<<<dim3(1024), TPB, 0, stream>>>(ypred, w_gt, part);
  k_loss_final<<<dim3(1), TPB, 0, stream>>>(part, out_kl, out_loss);

  // final primal update with y_pred as p1
  k_final_w<<<gBL, TPB, 0, stream>>>(w, y1, p2, ypred, z, ei, ej, gn, beta, out_wlist);
}
